// ShakesGPT_12652973654568
// MI455X (gfx1250) — matrix operand order repaired and checked by reading, not compiled, not run on hardware
//
#include <hip/hip_runtime.h>

typedef long long i64;
typedef unsigned short bf16u;
typedef __attribute__((ext_vector_type(16))) __bf16 v16bf;
typedef __attribute__((ext_vector_type(8)))  float  v8f;
typedef __attribute__((ext_vector_type(4)))  unsigned int u32x4;

union Frag16 { u32x4 u[2]; v16bf v; };
static_assert(sizeof(Frag16) == 32, "frag size");

static __device__ __forceinline__ bf16u f2bf(float f) {
  unsigned int u = __float_as_uint(f);
  u += 0x7FFFu + ((u >> 16) & 1u);   // round-to-nearest-even
  return (bf16u)(u >> 16);
}

static __device__ __forceinline__ v8f wmma_bf16(v16bf a, v16bf b, v8f c) {
  return __builtin_amdgcn_wmma_f32_16x16x32_bf16(false, a, false, b, (short)0, c, false, false);
}

// CDNA5 async memory->LDS copy (16B per lane), tracked by ASYNCcnt.
// lds_off = wave-relative LDS byte address (low 32 bits of generic pointer).
static __device__ __forceinline__ void async_ld_b128(unsigned lds_off, const void* gaddr) {
  asm volatile("global_load_async_to_lds_b128 %0, %1, off"
               :: "v"(lds_off), "v"(gaddr) : "memory");
}
static __device__ __forceinline__ void async_ld_b128_off16(unsigned lds_off, const void* gaddr) {
  // 24-bit inst offset applies to BOTH the LDS and global address.
  asm volatile("global_load_async_to_lds_b128 %0, %1, off offset:16"
               :: "v"(lds_off), "v"(gaddr) : "memory");
}
static __device__ __forceinline__ void wait_async0() {
  asm volatile("s_wait_asynccnt 0x0" ::: "memory");
}
static __device__ __forceinline__ unsigned lds_off32(const void* p) {
  return (unsigned)(size_t)p;   // flat-LDS aperture: addr[31:0] = LDS offset
}

// ---------------------------------------------------------------------------
// Embedding: x[b,t,:] = tok_emb[idx[b,t],:] + pos_emb[t,:]
// ---------------------------------------------------------------------------
__global__ __launch_bounds__(256) void embed_kernel(
    const int* __restrict__ idx, const float* __restrict__ tok,
    const float* __restrict__ pos, float* __restrict__ x, int M, int T, int E) {
  i64 i = (i64)blockIdx.x * 256 + threadIdx.x;
  i64 total = (i64)M * E;
  if (i >= total) return;
  int row = (int)(i / E), col = (int)(i % E);
  int t = row % T;
  x[i] = tok[(i64)idx[row] * E + col] + pos[(i64)t * E + col];
}

// ---------------------------------------------------------------------------
// f32 [K,N] -> bf16 [N,K]  (weight convert + transpose for WMMA B-operand)
// ---------------------------------------------------------------------------
__global__ __launch_bounds__(256) void convtrans_kernel(
    const float* __restrict__ in, bf16u* __restrict__ out, int K, int N) {
  i64 i = (i64)blockIdx.x * 256 + threadIdx.x;
  i64 total = (i64)K * N;
  if (i >= total) return;
  i64 k = i / N, n = i % N;
  out[n * (i64)K + k] = f2bf(in[i]);
}

// ---------------------------------------------------------------------------
// LayerNorm over E per row; bf16 output (feeds GEMM A operand)
// ---------------------------------------------------------------------------
__global__ __launch_bounds__(256) void ln_kernel(
    const float* __restrict__ x, const float* __restrict__ g,
    const float* __restrict__ b, bf16u* __restrict__ out, int E) {
  const int row = blockIdx.x;
  const float* xr = x + (i64)row * E;
  __shared__ float red[256];
  const int tid = threadIdx.x;

  float s = 0.f;
  for (int c = tid; c < E; c += 256) s += xr[c];
  red[tid] = s; __syncthreads();
  for (int o = 128; o > 0; o >>= 1) { if (tid < o) red[tid] += red[tid + o]; __syncthreads(); }
  float mu = red[0] / (float)E;
  __syncthreads();

  float v = 0.f;
  for (int c = tid; c < E; c += 256) { float d = xr[c] - mu; v += d * d; }
  red[tid] = v; __syncthreads();
  for (int o = 128; o > 0; o >>= 1) { if (tid < o) red[tid] += red[tid + o]; __syncthreads(); }
  float rsig = rsqrtf(red[0] / (float)E + 1e-5f);

  for (int c = tid; c < E; c += 256)
    out[(i64)row * E + c] = f2bf((xr[c] - mu) * rsig * g[c] + b[c]);
}

// ---------------------------------------------------------------------------
// WMMA GEMM: C[M,N] = A_bf16[M,K] @ Bt_bf16[N,K]^T  (+bias, +residual, relu)
// Block tile 64x128, 8 waves of 32x32 (2x2 WMMA 16x16x32), BK=32.
// Double-buffered LDS fed by global_load_async_to_lds_b128 (ASYNCcnt);
// one barrier per K-step, async copy of tile i+1 overlaps WMMAs of tile i.
// ---------------------------------------------------------------------------
__global__ __launch_bounds__(256) void gemm_bf16_kernel(
    const bf16u* __restrict__ A, const bf16u* __restrict__ Bt,
    const float* __restrict__ bias, const float* __restrict__ res,
    float* __restrict__ Cf, bf16u* __restrict__ Cb,
    int M, int N, int K, int relu) {
  __shared__ bf16u As[2][64][32];   // [buf][m][k]
  __shared__ bf16u Bs[2][128][32];  // [buf][n][k]

  const int tid  = threadIdx.x;
  const int w    = tid >> 5;
  const int lane = tid & 31;
  const int h    = lane >> 4;    // 16-lane half
  const int l16  = lane & 15;
  const int wm   = (w & 1) * 32;
  const int wn   = (w >> 1) * 32;
  const i64 m0 = (i64)blockIdx.y * 64;
  const i64 n0 = (i64)blockIdx.x * 128;

  const int ar = tid >> 2, ac = (tid & 3) * 8;   // A tile: 64 rows x 32, 8 elems/thread
  const int br = tid >> 1, bc = (tid & 1) * 16;  // B tile: 128 rows x 32, 16 elems/thread

  auto stage = [&](int buf, int k0) {
    async_ld_b128(lds_off32(&As[buf][ar][ac]), &A[(m0 + ar) * (i64)K + k0 + ac]);
    const bf16u* gb = &Bt[(n0 + br) * (i64)K + k0 + bc];
    unsigned lb = lds_off32(&Bs[buf][br][bc]);
    async_ld_b128(lb, gb);          // 8 elems
    async_ld_b128_off16(lb, gb);    // next 8 elems (offset applies both sides)
  };

  v8f z = {0.f,0.f,0.f,0.f,0.f,0.f,0.f,0.f};
  v8f acc[2][2] = {{z, z}, {z, z}};

  const int nIter = K >> 5;
  stage(0, 0);
  for (int it = 0; it < nIter; ++it) {
    const int buf = it & 1;
    wait_async0();        // my async copies for tile `it` are in LDS
    __syncthreads();      // everyone's are; everyone done reading buf^1
    if (it + 1 < nIter) stage(buf ^ 1, (it + 1) << 5);

    Frag16 af[2], bfr[2];
#pragma unroll
    for (int mi = 0; mi < 2; ++mi) {
      int r = wm + mi * 16 + l16;
      af[mi].u[0] = *(const u32x4*)&As[buf][r][h * 8];        // K = h*8 + 0..7
      af[mi].u[1] = *(const u32x4*)&As[buf][r][16 + h * 8];   // K = 16 + h*8 + 0..7
    }
#pragma unroll
    for (int ni = 0; ni < 2; ++ni) {
      int c = wn + ni * 16 + l16;
      bfr[ni].u[0] = *(const u32x4*)&Bs[buf][c][h * 8];       // K = h*8 + 0..7
      bfr[ni].u[1] = *(const u32x4*)&Bs[buf][c][16 + h * 8];  // K = 16 + h*8 + 0..7
    }
#pragma unroll
    for (int mi = 0; mi < 2; ++mi)
#pragma unroll
      for (int ni = 0; ni < 2; ++ni)
        acc[mi][ni] = wmma_bf16(af[mi].v, bfr[ni].v, acc[mi][ni]);
    // no trailing barrier: frag ds_loads are consumed (dscnt-waited) before
    // the next iteration's barrier, after which buf is safe to overwrite.
  }

  // Epilogue: C layout -> lane l holds rows h*8+r, col l16 of each 16x16 tile.
#pragma unroll
  for (int mi = 0; mi < 2; ++mi) {
#pragma unroll
    for (int ni = 0; ni < 2; ++ni) {
      i64 col = n0 + wn + ni * 16 + l16;
      float bv = bias ? bias[col] : 0.f;
#pragma unroll
      for (int r = 0; r < 8; ++r) {
        i64 row = m0 + wm + mi * 16 + h * 8 + r;
        i64 o = row * (i64)N + col;
        float v = acc[mi][ni][r] + bv;
        if (res)  v += res[o];
        if (relu) v = fmaxf(v, 0.f);
        if (Cf) Cf[o] = v;
        if (Cb) Cb[o] = f2bf(v);
      }
    }
  }
}

// ---------------------------------------------------------------------------
// Flash attention (causal), WMMA. One workgroup = 128 q rows of one (b,h);
// each wave owns 16 q rows. 32-key steps: K async-staged to LDS, V staged
// transposed via VGPRs, online softmax with 16-lane shuffle reductions,
// P re-laid out C->A via per-wave LDS scratch.
// ---------------------------------------------------------------------------
__global__ __launch_bounds__(256) void attn_kernel(
    const bf16u* __restrict__ Qb, const bf16u* __restrict__ Kb,
    const bf16u* __restrict__ Vb, bf16u* __restrict__ Ob,
    int T, int E, int H) {
  __shared__ bf16u Ks[32][64];   // [key][d]
  __shared__ bf16u Vt[64][40];   // [d][key], padded rows (80B stride)
  __shared__ bf16u Pl[8][16 * 32]; // per-wave P scratch

  const int qblk = blockIdx.x * 128;
  const int bb = blockIdx.y / H, hh = blockIdx.y % H;
  const int tid = threadIdx.x, w = tid >> 5, lane = tid & 31;
  const int h = lane >> 4, l16 = lane & 15;
  const float scale = 0.125f;  // D^-0.5, D=64
  const i64 base = ((i64)bb * T) * E + (i64)hh * 64;

  // Q A-fragments for this wave's 16 rows (d split in two 32-chunks)
  Frag16 qf[2];
  {
    i64 qoff = base + (i64)(qblk + w * 16 + l16) * E;
#pragma unroll
    for (int ks = 0; ks < 2; ++ks) {
      qf[ks].u[0] = *(const u32x4*)&Qb[qoff + ks * 32 + h * 8];
      qf[ks].u[1] = *(const u32x4*)&Qb[qoff + ks * 32 + 16 + h * 8];
    }
  }

  v8f z = {0.f,0.f,0.f,0.f,0.f,0.f,0.f,0.f};
  v8f oacc[4] = {z, z, z, z};
  float rm[8], rs[8];
#pragma unroll
  for (int r = 0; r < 8; ++r) { rm[r] = -1e30f; rs[r] = 0.f; }

  const int qr0 = qblk + w * 16 + h * 8;  // + r = this lane's rows
  const int kend = qblk + 128;
  const int kr = tid >> 3, kc = (tid & 7) * 8;

  for (int kb = 0; kb < kend; kb += 32) {
    // stage K (async, raw) and V (transposed, via VGPRs)
    async_ld_b128(lds_off32(&Ks[kr][kc]), &Kb[base + (i64)(kb + kr) * E + kc]);
    {
      union { u32x4 u; bf16u s[8]; } tv;
      tv.u = *(const u32x4*)&Vb[base + (i64)(kb + kr) * E + kc];
#pragma unroll
      for (int j = 0; j < 8; ++j) Vt[kc + j][kr] = tv.s[j];
    }
    wait_async0();
    __syncthreads();

    // scores: two 16x16 tiles (keys kb..kb+15, kb+16..kb+31)
    v8f s0 = z, s1 = z;
#pragma unroll
    for (int ks = 0; ks < 2; ++ks) {
      Frag16 b0, b1;
      b0.u[0] = *(const u32x4*)&Ks[l16][ks * 32 + h * 8];
      b0.u[1] = *(const u32x4*)&Ks[l16][ks * 32 + 16 + h * 8];
      b1.u[0] = *(const u32x4*)&Ks[16 + l16][ks * 32 + h * 8];
      b1.u[1] = *(const u32x4*)&Ks[16 + l16][ks * 32 + 16 + h * 8];
      s0 = wmma_bf16(qf[ks].v, b0.v, s0);
      s1 = wmma_bf16(qf[ks].v, b1.v, s1);
    }

    // online softmax (per row qr0+r; 16 lanes of same h share a row)
    float p0[8], p1[8];
#pragma unroll
    for (int r = 0; r < 8; ++r) {
      float a = s0[r] * scale;
      float c = s1[r] * scale;
      if (kb + l16      > qr0 + r) a = -1e30f;
      if (kb + 16 + l16 > qr0 + r) c = -1e30f;
      float mx = fmaxf(a, c);
#pragma unroll
      for (int d = 1; d < 16; d <<= 1) mx = fmaxf(mx, __shfl_xor(mx, d, 32));
      float mnew = fmaxf(rm[r], mx);
      float corr = __expf(rm[r] - mnew);
      float e0 = __expf(a - mnew), e1 = __expf(c - mnew);
      float ss = e0 + e1;
#pragma unroll
      for (int d = 1; d < 16; d <<= 1) ss += __shfl_xor(ss, d, 32);
      rs[r] = rs[r] * corr + ss;
      rm[r] = mnew;
      p0[r] = e0; p1[r] = e1;
#pragma unroll
      for (int d = 0; d < 4; ++d) oacc[d][r] *= corr;
    }

    // re-layout P: C layout -> LDS -> A layout (per-wave scratch, DS in-order)
    bf16u* pw = &Pl[w][0];
#pragma unroll
    for (int r = 0; r < 8; ++r) {
      pw[(h * 8 + r) * 32 + l16]      = f2bf(p0[r]);
      pw[(h * 8 + r) * 32 + 16 + l16] = f2bf(p1[r]);
    }
    Frag16 pf;
    pf.u[0] = *(const u32x4*)&pw[l16 * 32 + h * 8];
    pf.u[1] = *(const u32x4*)&pw[l16 * 32 + 16 + h * 8];

    // O += P (16x32) @ V (32x64): 4 d-subtiles
#pragma unroll
    for (int d = 0; d < 4; ++d) {
      Frag16 vf;
      vf.u[0] = *(const u32x4*)&Vt[d * 16 + l16][h * 8];
      vf.u[1] = *(const u32x4*)&Vt[d * 16 + l16][16 + h * 8];
      oacc[d] = wmma_bf16(pf.v, vf.v, oacc[d]);
    }
    __syncthreads();
  }

  // normalize + store bf16 O
#pragma unroll
  for (int r = 0; r < 8; ++r) {
    float inv = 1.0f / rs[r];
    i64 off = base + (i64)(qr0 + r) * E;
#pragma unroll
    for (int d = 0; d < 4; ++d)
      Ob[off + d * 16 + l16] = f2bf(oacc[d][r] * inv);
  }
}

// ---------------------------------------------------------------------------
// Host launcher
// ---------------------------------------------------------------------------
static void launch_gemm(const bf16u* A, const bf16u* Bt, const float* bias,
                        const float* res, float* Cf, bf16u* Cb,
                        int M, int N, int K, int relu, hipStream_t s) {
  dim3 g(N / 128, M / 64);
  gemm_bf16_kernel<<<g, 256, 0, s>>>(A, Bt, bias, res, Cf, Cb, M, N, K, relu);
}

static void launch_convtrans(const float* in, bf16u* out, int K, int N, hipStream_t s) {
  i64 total = (i64)K * N;
  int blocks = (int)((total + 255) / 256);
  convtrans_kernel<<<blocks, 256, 0, s>>>(in, out, K, N);
}

extern "C" void kernel_launch(void* const* d_in, const int* in_sizes, int n_in,
                              void* d_out, int out_size, void* d_ws, size_t ws_size,
                              hipStream_t stream) {
  (void)in_sizes; (void)n_in; (void)out_size; (void)ws_size;
  const int V = 32000, E = 1024, H = 16, L = 6, FF = 4096, B = 2, T = 1024;
  const int M = B * T;

  const int*   idx  = (const int*)  d_in[0];
  const float* tok  = (const float*)d_in[1];
  const float* pos  = (const float*)d_in[2];
  const float* Wq   = (const float*)d_in[3];
  const float* Wk   = (const float*)d_in[4];
  const float* Wv   = (const float*)d_in[5];
  const float* Wp   = (const float*)d_in[6];
  const float* bp   = (const float*)d_in[7];
  const float* g1   = (const float*)d_in[8];
  const float* bb1  = (const float*)d_in[9];
  const float* g2   = (const float*)d_in[10];
  const float* bb2  = (const float*)d_in[11];
  const float* W1   = (const float*)d_in[12];
  const float* bf1  = (const float*)d_in[13];
  const float* W2   = (const float*)d_in[14];
  const float* bf2  = (const float*)d_in[15];
  const float* gf   = (const float*)d_in[16];
  const float* bfb  = (const float*)d_in[17];
  const float* Wlm  = (const float*)d_in[18];
  const float* blm  = (const float*)d_in[19];

  char* wsp = (char*)d_ws;
  size_t off = 0;
  auto alloc = [&](size_t bytes) -> void* {
    void* p = wsp + off;
    off += (bytes + 255) & ~(size_t)255;
    return p;
  };
  float* x    = (float*)alloc((size_t)M * E * 4);
  bf16u* hbf  = (bf16u*)alloc((size_t)M * E * 2);
  bf16u* qbf  = (bf16u*)alloc((size_t)M * E * 2);
  bf16u* kbf  = (bf16u*)alloc((size_t)M * E * 2);
  bf16u* vbf  = (bf16u*)alloc((size_t)M * E * 2);
  bf16u* obf  = (bf16u*)alloc((size_t)M * E * 2);
  bf16u* ffbf = (bf16u*)alloc((size_t)M * FF * 2);
  bf16u* wt   = (bf16u*)alloc((size_t)E * V * 2);  // reused for every weight

  // embeddings
  {
    i64 total = (i64)M * E;
    embed_kernel<<<(int)((total + 255) / 256), 256, 0, stream>>>(idx, tok, pos, x, M, T, E);
  }

  for (int l = 0; l < L; ++l) {
    const i64 lEE = (i64)l * E * E;
    // LN1
    ln_kernel<<<M, 256, 0, stream>>>(x, g1 + (i64)l * E, bb1 + (i64)l * E, hbf, E);
    // Q, K, V projections (no bias)
    launch_convtrans(Wq + lEE, wt, E, E, stream);
    launch_gemm(hbf, wt, nullptr, nullptr, nullptr, qbf, M, E, E, 0, stream);
    launch_convtrans(Wk + lEE, wt, E, E, stream);
    launch_gemm(hbf, wt, nullptr, nullptr, nullptr, kbf, M, E, E, 0, stream);
    launch_convtrans(Wv + lEE, wt, E, E, stream);
    launch_gemm(hbf, wt, nullptr, nullptr, nullptr, vbf, M, E, E, 0, stream);
    // causal flash attention
    attn_kernel<<<dim3(T / 128, B * H), 256, 0, stream>>>(qbf, kbf, vbf, obf, T, E, H);
    // x = x + O @ Wp + bp
    launch_convtrans(Wp + lEE, wt, E, E, stream);
    launch_gemm(obf, wt, bp + (i64)l * E, x, x, nullptr, M, E, E, 0, stream);
    // LN2
    ln_kernel<<<M, 256, 0, stream>>>(x, g2 + (i64)l * E, bb2 + (i64)l * E, hbf, E);
    // ff = relu(h2 @ W1 + b1)
    launch_convtrans(W1 + (i64)l * E * FF, wt, E, FF, stream);
    launch_gemm(hbf, wt, bf1 + (i64)l * FF, nullptr, nullptr, ffbf, M, FF, E, 1, stream);
    // x = x + ff @ W2 + b2
    launch_convtrans(W2 + (i64)l * FF * E, wt, FF, E, stream);
    launch_gemm(ffbf, wt, bf2 + (i64)l * E, x, x, nullptr, M, E, FF, 0, stream);
  }

  // final LN + LM head
  ln_kernel<<<M, 256, 0, stream>>>(x, gf, bfb, hbf, E);
  launch_convtrans(Wlm, wt, E, V, stream);
  launch_gemm(hbf, wt, blm, nullptr, (float*)d_out, nullptr, M, V, E, 0, stream);
}
